// AttentionLayer_52956946760482
// MI455X (gfx1250) — matrix operand order repaired and checked by reading, not compiled, not run on hardware
//
#include <hip/hip_runtime.h>
#include <stdint.h>

#define DEV __device__ __forceinline__

typedef __attribute__((ext_vector_type(16))) __bf16 v16bf;
typedef __attribute__((ext_vector_type(2)))  __bf16 v2bf;
typedef __attribute__((ext_vector_type(8)))  float  v8f;

union BF8x32 { uint32_t u[8]; v16bf v; };   // 8 dwords = 16 packed bf16

// fp32 -> bf16 using the hardware converter (packed when available).
DEV uint32_t pack_bf16(float lo, float hi) {
#if __has_builtin(__builtin_amdgcn_cvt_pk_bf16_f32)
  v2bf p = __builtin_amdgcn_cvt_pk_bf16_f32(lo, hi);
  return __builtin_bit_cast(uint32_t, p);
#else
  v2bf p;
  p.x = (__bf16)lo;
  p.y = (__bf16)hi;
  return __builtin_bit_cast(uint32_t, p);
#endif
}
DEV uint16_t f2bf(float f) {
  __bf16 b = (__bf16)f;
  return __builtin_bit_cast(uint16_t, b);
}

// A-matrix (16x32, 16-bit) K offset for vgpr vi, lane-half: per ISA 7.12.2
DEV int a_kofs(int vi, int half) { return ((vi & 4) << 2) + half * 8 + ((vi & 3) << 1); }
// B-matrix (32x16, 16-bit): same K order as the A-matrix
DEV int b_kofs(int vi, int half) { return ((vi & 4) << 2) + half * 8 + ((vi & 3) << 1); }

DEV v8f wmma_bf16(v16bf a, v16bf b, v8f c) {
  return __builtin_amdgcn_wmma_f32_16x16x32_bf16(false, a, false, b, (short)0, c,
                                                 false, false);
}

// ---------------------------------------------------------------------------
// Kernel 1: Q/K/V projections.  out = X @ W + bias, stored bf16.
// grid (64, 8, 3), block 256 (8 waves); one 16x16 tile per wave.
// ---------------------------------------------------------------------------
__global__ __launch_bounds__(256) void qkv_proj_kernel(
    const float* __restrict__ Xq, const float* __restrict__ Xk, const float* __restrict__ Xv,
    const float* __restrict__ Wq, const float* __restrict__ Wk, const float* __restrict__ Wv,
    const float* __restrict__ bq, const float* __restrict__ bk, const float* __restrict__ bv,
    uint16_t* __restrict__ Oq, uint16_t* __restrict__ Ok, uint16_t* __restrict__ Ov)
{
  const int D = 128;
  const int z = blockIdx.z;
  const float* X    = (z == 0) ? Xq : (z == 1) ? Xk : Xv;
  const float* W    = (z == 0) ? Wq : (z == 1) ? Wk : Wv;
  const float* bias = (z == 0) ? bq : (z == 1) ? bk : bv;
  uint16_t*    out  = (z == 0) ? Oq : (z == 1) ? Ok : Ov;

  const int lane = threadIdx.x & 31, wv = threadIdx.x >> 5;
  const int lm = lane & 15, half = lane >> 4;
  const int rt = blockIdx.x * 8 + wv;     // row tile (0..511), 8192 rows total
  const int ct = blockIdx.y;              // col tile (0..7)
  const int mrow = rt * 16 + lm;
  const int ncol = ct * 16 + lm;

  v8f acc = {0.f, 0.f, 0.f, 0.f, 0.f, 0.f, 0.f, 0.f};
#pragma unroll
  for (int kk = 0; kk < 128; kk += 32) {
    BF8x32 af, bfr;
#pragma unroll
    for (int vi = 0; vi < 8; ++vi) {
      const int ka = kk + a_kofs(vi, half);
      const float* xp = X + (size_t)mrow * D + ka;
      af.u[vi] = pack_bf16(xp[0], xp[1]);
      const int kb = kk + b_kofs(vi, half);
      bfr.u[vi] = pack_bf16(W[(size_t)kb * D + ncol], W[(size_t)(kb + 1) * D + ncol]);
    }
    acc = wmma_bf16(af.v, bfr.v, acc);
  }
  const float bb = bias[ncol];
#pragma unroll
  for (int r = 0; r < 8; ++r) {
    const int mg = rt * 16 + r + 8 * half;
    out[(size_t)mg * D + ncol] = f2bf(acc[r] + bb);
  }
}

// ---------------------------------------------------------------------------
// Kernel 2: fused dual-softmax flash attention.
// grid (B*H=64, N/128=8), block 256 (8 waves); each wave owns 16 query rows.
// Per 32-key step: 2 QK^T WMMAs, online softmax (global + adjacency-masked),
// P re-layout C-tile -> A-fragment via LDS, 2 P*V WMMAs.
// ---------------------------------------------------------------------------
__global__ __launch_bounds__(256) void attn_kernel(
    const uint16_t* __restrict__ Qbf, const uint16_t* __restrict__ Kbf,
    const uint16_t* __restrict__ Vbf, const uint8_t* __restrict__ adj,
    uint16_t* __restrict__ AttOut)
{
  const int N = 1024, D = 128;
  const int lane = threadIdx.x & 31, wv = threadIdx.x >> 5;
  const int lm = lane & 15, half = lane >> 4;
  const int b = blockIdx.x >> 3, h = blockIdx.x & 7;
  const int q0 = blockIdx.y * 128 + wv * 16;

  __shared__ uint16_t lds_p[8][2][16 * 32];   // per-wave P staging (16KB)

  // Q A-fragment: K = d_h = 16, zero-padded to 32 (v4..7 are the K=16..31 pads)
  BF8x32 qf;
  {
    const uint32_t* qrow =
        (const uint32_t*)(Qbf + ((size_t)(b * N + q0 + lm)) * D + h * 16);
#pragma unroll
    for (int vi = 0; vi < 4; ++vi) qf.u[vi] = qrow[half * 4 + vi];
#pragma unroll
    for (int vi = 4; vi < 8; ++vi) qf.u[vi] = 0u;
  }

  const float NEG_INF = -__builtin_inff();
  float mg[8], lg[8], ml[8], ll[8];
  v8f og = {0.f, 0.f, 0.f, 0.f, 0.f, 0.f, 0.f, 0.f};
  v8f ol = {0.f, 0.f, 0.f, 0.f, 0.f, 0.f, 0.f, 0.f};
  const v8f zacc = {0.f, 0.f, 0.f, 0.f, 0.f, 0.f, 0.f, 0.f};
#pragma unroll
  for (int r = 0; r < 8; ++r) { mg[r] = ml[r] = NEG_INF; lg[r] = ll[r] = 0.f; }

#pragma unroll 1
  for (int kb = 0; kb < N; kb += 32) {
    // K B-fragments for the two 16-key subtiles (B-matrix K dim = d_h pad 32):
    // v0..3: lanes 0-15 carry d_h 0..7, lanes 16-31 carry d_h 8..15; v4..7 zero pad
    BF8x32 kf0, kf1;
    {
      const uint32_t* k0p =
          (const uint32_t*)(Kbf + ((size_t)(b * N + kb + lm)) * D + h * 16);
      const uint32_t* k1p =
          (const uint32_t*)(Kbf + ((size_t)(b * N + kb + 16 + lm)) * D + h * 16);
#pragma unroll
      for (int vi = 0; vi < 8; ++vi) {
        kf0.u[vi] = (vi < 4) ? k0p[half * 4 + vi] : 0u;
        kf1.u[vi] = (vi < 4) ? k1p[half * 4 + vi] : 0u;
      }
    }
    // V B-fragment (K dim = 32 keys, N dim = 16 head channels)
    BF8x32 vf;
#pragma unroll
    for (int vi = 0; vi < 8; ++vi) {
      const int kk = half * 16 + 2 * vi;
      const uint16_t* vp = Vbf + ((size_t)(b * N + kb + kk)) * D + h * 16 + lm;
      vf.u[vi] = (uint32_t)vp[0] | ((uint32_t)vp[D] << 16);
    }
    if (kb + 32 < N) {   // prefetch next K/V tile rows (global_prefetch_b8)
      __builtin_prefetch(Kbf + ((size_t)(b * N + kb + 32 + lm)) * D + h * 16, 0, 3);
      __builtin_prefetch(Vbf + ((size_t)(b * N + kb + 32 + lm)) * D + h * 16, 0, 3);
    }

    v8f s0 = wmma_bf16(qf.v, kf0.v, zacc);
    v8f s1 = wmma_bf16(qf.v, kf1.v, zacc);

    float p0g[8], p1g[8], ag[8];
    float p0l[8], p1l[8], al[8];
#pragma unroll
    for (int r = 0; r < 8; ++r) {
      const int mrow = q0 + r + 8 * half;                 // C layout: M = r + 8*half
      const float sg0 = s0[r] * 0.25f;                    // 1/sqrt(d_h)
      const float sg1 = s1[r] * 0.25f;
      const uint8_t a0 = adj[(size_t)mrow * N + kb + lm];
      const uint8_t a1 = adj[(size_t)mrow * N + kb + 16 + lm];
      const float sl0 = a0 ? sg0 : NEG_INF;
      const float sl1 = a1 ? sg1 : NEG_INF;
      { // global softmax, online update (row reduce across the 16-lane group)
        float t = fmaxf(sg0, sg1);
        t = fmaxf(t, __shfl_xor(t, 1, 16));
        t = fmaxf(t, __shfl_xor(t, 2, 16));
        t = fmaxf(t, __shfl_xor(t, 4, 16));
        t = fmaxf(t, __shfl_xor(t, 8, 16));
        const float mnew = fmaxf(mg[r], t);
        const float a = (mg[r] == NEG_INF) ? 0.f : __expf(mg[r] - mnew);
        const float e0 = __expf(sg0 - mnew);
        const float e1 = __expf(sg1 - mnew);
        float rs = e0 + e1;
        rs += __shfl_xor(rs, 1, 16);
        rs += __shfl_xor(rs, 2, 16);
        rs += __shfl_xor(rs, 4, 16);
        rs += __shfl_xor(rs, 8, 16);
        lg[r] = a * lg[r] + rs;  mg[r] = mnew;  ag[r] = a;
        p0g[r] = e0;  p1g[r] = e1;
      }
      { // adjacency-masked softmax
        float t = fmaxf(sl0, sl1);
        t = fmaxf(t, __shfl_xor(t, 1, 16));
        t = fmaxf(t, __shfl_xor(t, 2, 16));
        t = fmaxf(t, __shfl_xor(t, 4, 16));
        t = fmaxf(t, __shfl_xor(t, 8, 16));
        const float mnew = fmaxf(ml[r], t);
        const float a = (ml[r] == NEG_INF) ? 0.f : __expf(ml[r] - mnew);
        const float e0 = (sl0 == NEG_INF) ? 0.f : __expf(sl0 - mnew);
        const float e1 = (sl1 == NEG_INF) ? 0.f : __expf(sl1 - mnew);
        float rs = e0 + e1;
        rs += __shfl_xor(rs, 1, 16);
        rs += __shfl_xor(rs, 2, 16);
        rs += __shfl_xor(rs, 4, 16);
        rs += __shfl_xor(rs, 8, 16);
        ll[r] = a * ll[r] + rs;  ml[r] = mnew;  al[r] = a;
        p0l[r] = e0;  p1l[r] = e1;
      }
    }

    // rescale accumulators, stage P (bf16) to LDS in C layout (packed pairs
    // are written as two b16 stores; conversion uses the packed hw cvt)
#pragma unroll
    for (int r = 0; r < 8; ++r) {
      og[r] *= ag[r];
      ol[r] *= al[r];
      const int mr = r + 8 * half;
      const uint32_t pgp = pack_bf16(p0g[r], p1g[r]);
      const uint32_t plp = pack_bf16(p0l[r], p1l[r]);
      lds_p[wv][0][mr * 32 + lm]      = (uint16_t)pgp;
      lds_p[wv][0][mr * 32 + 16 + lm] = (uint16_t)(pgp >> 16);
      lds_p[wv][1][mr * 32 + lm]      = (uint16_t)plp;
      lds_p[wv][1][mr * 32 + 16 + lm] = (uint16_t)(plp >> 16);
    }
    asm volatile("s_wait_dscnt 0" ::: "memory");   // same-wave LDS store->load

    // reload P as A-fragments (16x32 bf16) and accumulate P @ V
    BF8x32 pg, pl;
    {
      const uint32_t* pgrow = (const uint32_t*)&lds_p[wv][0][lm * 32];
      const uint32_t* plrow = (const uint32_t*)&lds_p[wv][1][lm * 32];
#pragma unroll
      for (int vi = 0; vi < 8; ++vi) {
        const int ka = half * 8 + vi;
        pg.u[vi] = pgrow[ka];
        pl.u[vi] = plrow[ka];
      }
    }
    og = wmma_bf16(pg.v, vf.v, og);
    ol = wmma_bf16(pl.v, vf.v, ol);
  }

  // epilogue: out = 0.5 * (Og/lg + Ol/ll), bf16 to workspace
#pragma unroll
  for (int r = 0; r < 8; ++r) {
    const int mr = q0 + r + 8 * half;
    const float vgl = og[r] / lg[r];
    const float vlc = (ll[r] > 0.f) ? (ol[r] / ll[r]) : 0.f;
    AttOut[((size_t)(b * N + mr)) * D + h * 16 + lm] = f2bf(0.5f * (vgl + vlc));
  }
}

// ---------------------------------------------------------------------------
// Kernel 3: output projection.  out(f32) = AttOut(bf16) @ Wo + bo.
// grid (64, 8), block 256; one 16x16 tile per wave.
// ---------------------------------------------------------------------------
__global__ __launch_bounds__(256) void out_proj_kernel(
    const uint16_t* __restrict__ Xbf, const float* __restrict__ Wo,
    const float* __restrict__ bo, float* __restrict__ out)
{
  const int D = 128;
  const int lane = threadIdx.x & 31, wv = threadIdx.x >> 5;
  const int lm = lane & 15, half = lane >> 4;
  const int rt = blockIdx.x * 8 + wv;
  const int ct = blockIdx.y;
  const int ncol = ct * 16 + lm;

  v8f acc = {0.f, 0.f, 0.f, 0.f, 0.f, 0.f, 0.f, 0.f};
  const uint32_t* xrow = (const uint32_t*)(Xbf + (size_t)(rt * 16 + lm) * D);
#pragma unroll
  for (int kk = 0; kk < 128; kk += 32) {
    BF8x32 af, bfr;
#pragma unroll
    for (int vi = 0; vi < 8; ++vi) {
      const int ka = kk + a_kofs(vi, half);
      af.u[vi] = xrow[ka >> 1];
      const int kb = kk + b_kofs(vi, half);
      bfr.u[vi] = pack_bf16(Wo[(size_t)kb * D + ncol], Wo[(size_t)(kb + 1) * D + ncol]);
    }
    acc = wmma_bf16(af.v, bfr.v, acc);
  }
  const float bb = bo[ncol];
#pragma unroll
  for (int r = 0; r < 8; ++r) {
    out[(size_t)(rt * 16 + r + 8 * half) * D + ncol] = acc[r] + bb;
  }
}

// ---------------------------------------------------------------------------
extern "C" void kernel_launch(void* const* d_in, const int* in_sizes, int n_in,
                              void* d_out, int out_size, void* d_ws, size_t ws_size,
                              hipStream_t stream) {
  (void)in_sizes; (void)n_in; (void)out_size; (void)ws_size;
  const float*   query = (const float*)d_in[0];
  const float*   key   = (const float*)d_in[1];
  const float*   value = (const float*)d_in[2];
  const uint8_t* adj   = (const uint8_t*)d_in[3];
  const float*   Wq    = (const float*)d_in[4];
  const float*   bq    = (const float*)d_in[5];
  const float*   Wk    = (const float*)d_in[6];
  const float*   bk    = (const float*)d_in[7];
  const float*   Wv    = (const float*)d_in[8];
  const float*   bv    = (const float*)d_in[9];
  const float*   Wo    = (const float*)d_in[10];
  const float*   bo    = (const float*)d_in[11];
  float* out = (float*)d_out;

  const size_t elems = 8ull * 1024ull * 128ull;   // B*N*D
  uint16_t* Qbf = (uint16_t*)d_ws;                // 2 MB each, 8 MB total
  uint16_t* Kbf = Qbf + elems;
  uint16_t* Vbf = Kbf + elems;
  uint16_t* Abf = Vbf + elems;

  qkv_proj_kernel<<<dim3(64, 8, 3), 256, 0, stream>>>(
      query, key, value, Wq, Wk, Wv, bq, bk, bv, Qbf, Kbf, Vbf);
  attn_kernel<<<dim3(64, 8), 256, 0, stream>>>(Qbf, Kbf, Vbf, adj, Abf);
  out_proj_kernel<<<dim3(64, 8), 256, 0, stream>>>(Abf, Wo, bo, out);
}
